// SGCA_21938692948271
// MI455X (gfx1250) — matrix operand order repaired and checked by reading, not compiled, not run on hardware
//
#include <hip/hip_runtime.h>
#include <hip/hip_bf16.h>
#include <math.h>

typedef __bf16 bf16;
typedef __attribute__((ext_vector_type(16))) __bf16 bf16x16;
typedef __attribute__((ext_vector_type(8)))  __bf16 bf16x8;
typedef __attribute__((ext_vector_type(4)))  __bf16 bf16x4;
typedef __attribute__((ext_vector_type(8)))  float  floatx8;

#define NHEADS 8
#define DH 32

// ---- WMMA fragment index helpers (CDNA5 ISA 7.12.2 layouts, wave32) ----
__device__ __forceinline__ int a_kidx(int lane, int i) {
  int v = i >> 1, p = i & 1;
  int base = (lane < 16) ? 0 : 8;
  int grp  = (v < 4) ? 0 : 16;
  return grp + base + 2 * (v & 3) + p;
}
__device__ __forceinline__ int b_kidx(int lane, int i) {
  return ((i < 8) ? 0 : 8) + ((lane < 16) ? 0 : 8) + i;
}

// ---- DPP16 butterfly max over each 16-lane group (no LDS, no waits) ----
// XOR masks {1,2,7,15} span the 4-bit lane space; all are row-local DPP ctrls.
template <int CTRL>
__device__ __forceinline__ float dpp_max_step(float v) {
  int iv = __float_as_int(v);
  int pv = __builtin_amdgcn_update_dpp(iv, iv, CTRL, 0xf, 0xf, false);
  return fmaxf(v, __int_as_float(pv));
}
__device__ __forceinline__ float group16_max(float v) {
  v = dpp_max_step<0x0B1>(v);  // quad_perm [1,0,3,2]  (xor 1)
  v = dpp_max_step<0x04E>(v);  // quad_perm [2,3,0,1]  (xor 2)
  v = dpp_max_step<0x141>(v);  // row_half_mirror      (xor 7)
  v = dpp_max_step<0x140>(v);  // row_mirror           (xor 15)
  return v;
}

// ---------------- LayerNorm over channel dim of NCHW ----------------
__global__ __launch_bounds__(256) void sgca_layernorm_kernel(
    const float* __restrict__ x, const float* __restrict__ w,
    const float* __restrict__ bia, bf16* __restrict__ xn, int C, int HW) {
  int b = blockIdx.x;
  int p = blockIdx.y * blockDim.x + threadIdx.x;
  const float* xb = x + (size_t)b * C * HW + p;
  float s1 = 0.f, s2 = 0.f;
  for (int c = 0; c < C; ++c) {
    float v = xb[(size_t)c * HW];
    s1 += v; s2 += v * v;
  }
  float mean = s1 / C;
  float var  = s2 / C - mean * mean;
  float rstd = rsqrtf(var + 1e-6f);
  bf16* ob = xn + (size_t)b * C * HW + p;
  for (int c = 0; c < C; ++c) {
    float v = (xb[(size_t)c * HW] - mean) * rstd * w[c] + bia[c];
    ob[(size_t)c * HW] = (bf16)v;
  }
}

// ---------------- affinity row sums ----------------
__global__ __launch_bounds__(256) void sgca_rowsum_kernel(
    const float* __restrict__ aff, float* __restrict__ rs, int HW) {
  int row = blockIdx.x;  // b*NSP + m
  const float* a = aff + (size_t)row * HW;
  float s = 0.f;
  for (int i = threadIdx.x; i < HW; i += blockDim.x) s += a[i];
  for (int m = 16; m >= 1; m >>= 1) s += __shfl_xor(s, m, 32);
  __shared__ float partial[8];
  int w = threadIdx.x >> 5, lane = threadIdx.x & 31;
  if (lane == 0) partial[w] = s;
  __syncthreads();
  if (threadIdx.x == 0) {
    float t = 0.f;
    int nw = (int)(blockDim.x >> 5);
    for (int i = 0; i < nw; ++i) t += partial[i];
    rs[row] = t;
  }
}

// ---------------- generic bf16 WMMA GEMM ----------------
// C[b,m,n] = sum_k A[b,m,k] * B[b,k,n]   (A f32 cast->bf16, B bf16)
// BT: B stored [N][K] row-major. SCALE: multiply row m by 1/(rowscale[m]+1e-16)
template <bool BT, bool SCALE>
__global__ __launch_bounds__(256) void sgca_gemm_kernel(
    const float* __restrict__ A, const bf16* __restrict__ B,
    bf16* __restrict__ C, const float* __restrict__ rowscale, int rsStride,
    int K, int lda, int ldb, int ldc, long sA, long sB, long sC) {
  __shared__ bf16 As[128 * 40];   // pitch 40 halves -> 80B rows (16B aligned)
  __shared__ bf16 Bs[32 * 136];   // pitch 136 halves -> 272B rows (16B aligned)
  int b = blockIdx.z;
  int bm = blockIdx.x * 128, bn = blockIdx.y * 128;
  int t = threadIdx.x, lane = t & 31, w = t >> 5;
  int wm = w >> 1, wn = w & 1;  // 4x2 wave grid, 32x64 per wave
  floatx8 acc[2][4];
  for (int i = 0; i < 2; ++i)
    for (int j = 0; j < 4; ++j)
      for (int e = 0; e < 8; ++e) acc[i][j][e] = 0.f;
  const float* Ab = A + (size_t)b * sA;
  const bf16*  Bb = B + (size_t)b * sB;

  for (int kk = 0; kk < K; kk += 32) {
    // ---- A tile: thread covers row r=t>>1, 16 contiguous k ----
    {
      int r = t >> 1, c0 = (t & 1) * 16;
      const float4* ap = (const float4*)(Ab + (size_t)(bm + r) * lda + kk + c0);
      float4 a0 = ap[0], a1 = ap[1], a2 = ap[2], a3 = ap[3];
      bf16x8 lo, hi;
      lo[0] = (bf16)a0.x; lo[1] = (bf16)a0.y; lo[2] = (bf16)a0.z; lo[3] = (bf16)a0.w;
      lo[4] = (bf16)a1.x; lo[5] = (bf16)a1.y; lo[6] = (bf16)a1.z; lo[7] = (bf16)a1.w;
      hi[0] = (bf16)a2.x; hi[1] = (bf16)a2.y; hi[2] = (bf16)a2.z; hi[3] = (bf16)a2.w;
      hi[4] = (bf16)a3.x; hi[5] = (bf16)a3.y; hi[6] = (bf16)a3.z; hi[7] = (bf16)a3.w;
      *(bf16x8*)&As[r * 40 + c0]     = lo;
      *(bf16x8*)&As[r * 40 + c0 + 8] = hi;
    }
    // ---- B tile ----
    if (!BT) {
      int k = t >> 3, n0 = (t & 7) * 16;  // 16 contiguous n
      const uint4* bp = (const uint4*)(Bb + (size_t)(kk + k) * ldb + bn + n0);
      uint4 b0 = bp[0], b1 = bp[1];
      *(uint4*)&Bs[k * 136 + n0]     = b0;
      *(uint4*)&Bs[k * 136 + n0 + 8] = b1;
    } else {
      int n = t >> 1, k0 = (t & 1) * 16;  // 16 contiguous k, transpose into LDS
      const uint4* bp = (const uint4*)(Bb + (size_t)(bn + n) * ldb + kk + k0);
      uint4 b0 = bp[0], b1 = bp[1];
      bf16 tmp[16];
      *(uint4*)&tmp[0] = b0;
      *(uint4*)&tmp[8] = b1;
      for (int e = 0; e < 16; ++e) Bs[(k0 + e) * 136 + n] = tmp[e];
    }
    __syncthreads();

    bf16x16 af[2], bfr[4];
    for (int tm = 0; tm < 2; ++tm)
      for (int i = 0; i < 16; ++i)
        af[tm][i] = As[(wm * 32 + tm * 16 + (lane & 15)) * 40 + a_kidx(lane, i)];
    for (int tn = 0; tn < 4; ++tn)
      for (int i = 0; i < 16; ++i)
        bfr[tn][i] = Bs[b_kidx(lane, i) * 136 + wn * 64 + tn * 16 + (lane & 15)];
    for (int tm = 0; tm < 2; ++tm)
      for (int tn = 0; tn < 4; ++tn)
        acc[tm][tn] = __builtin_amdgcn_wmma_f32_16x16x32_bf16(
            false, af[tm], false, bfr[tn], (short)0, acc[tm][tn], false, false);
    __syncthreads();
  }

  int half = (lane < 16) ? 0 : 8;
  int col16 = lane & 15;
  for (int tm = 0; tm < 2; ++tm)
    for (int tn = 0; tn < 4; ++tn)
      for (int j = 0; j < 8; ++j) {
        int m = bm + wm * 32 + tm * 16 + j + half;
        int n = bn + wn * 64 + tn * 16 + col16;
        float v = acc[tm][tn][j];
        if (SCALE) v *= 1.f / (rowscale[b * rsStride + m] + 1e-16f);
        C[(size_t)b * sC + (size_t)m * ldc + n] = (bf16)v;
      }
}

// ---------------- strided flash attention (d = 32) ----------------
// O[m,d] = softmax_rows( scale * Q[m,:] @ K[n,:]^T ) @ V[n,d]
__global__ __launch_bounds__(256) void sgca_flash_kernel(
    const bf16* __restrict__ Q, const bf16* __restrict__ Kp,
    const bf16* __restrict__ Vp, bf16* __restrict__ Ob, float* __restrict__ Of,
    int Nk, float scale,
    long QB, long QH, int QM, int QD,
    long KB, long KH, int KN, int KD,
    long VB, long VH, int VN, int VD,
    long OB, long OH, int OM, int OD) {
  __shared__ bf16 klds[32 * 36];       // [n][d], 8B-aligned rows
  __shared__ bf16 vlds[32 * 36];
  __shared__ bf16 plds[8][16 * 34];    // wave-private P staging
  __shared__ float ostage[8][32 * 20]; // wave-private [d][m] f32 output transpose
  int bh = blockIdx.y;
  int b = bh >> 3, h = bh & 7;
  int t = threadIdx.x, lane = t & 31, w = t >> 5;
  const bf16* Qg = Q  + (size_t)b * QB + (size_t)h * QH;
  const bf16* Kg = Kp + (size_t)b * KB + (size_t)h * KH;
  const bf16* Vg = Vp + (size_t)b * VB + (size_t)h * VH;
  int m0 = blockIdx.x * 128 + w * 16;
  int mrow = m0 + (lane & 15);

  // Q A-fragment, scale folded in (two-phase: gather, then convert)
  bf16x16 qa;
  {
    const bf16* qp = Qg + (size_t)mrow * QM;
    bf16 qtmp[16];
    for (int i = 0; i < 16; ++i) qtmp[i] = qp[(size_t)a_kidx(lane, i) * QD];
    for (int i = 0; i < 16; ++i) qa[i] = (bf16)((float)qtmp[i] * scale);
  }
  bf16x16 ones;
  for (int i = 0; i < 16; ++i) ones[i] = (bf16)1.0f;

  floatx8 acc0, acc1, accS;  // accS: running softmax denominator via P @ ones
  for (int e = 0; e < 8; ++e) { acc0[e] = 0.f; acc1[e] = 0.f; accS[e] = 0.f; }
  float mrun[8];
  for (int j = 0; j < 8; ++j) mrun[j] = -INFINITY;

  for (int nb = 0; nb < Nk; nb += 32) {
    // cooperative K/V tile load: thread t -> key n=t>>3, 4 consecutive d
    {
      int n = t >> 3, d0 = (t & 7) * 4;
      const bf16* kp = Kg + (size_t)(nb + n) * KN + (size_t)d0 * KD;
      const bf16* vp = Vg + (size_t)(nb + n) * VN + (size_t)d0 * VD;
      bf16 k0 = kp[0], k1 = kp[KD], k2 = kp[2 * KD], k3 = kp[3 * KD];
      bf16 v0 = vp[0], v1 = vp[VD], v2 = vp[2 * VD], v3 = vp[3 * VD];
      bf16x4 kv = {k0, k1, k2, k3};
      bf16x4 vv = {v0, v1, v2, v3};
      *(bf16x4*)&klds[n * 36 + d0] = kv;
      *(bf16x4*)&vlds[n * 36 + d0] = vv;
    }
    __syncthreads();

    // logits: S[16m x 32n] = Q @ K^T (K-dim = d = 32)
    bf16x16 kb0, kb1;
    for (int i = 0; i < 16; ++i) {
      int kd = b_kidx(lane, i);
      kb0[i] = klds[(0  + (lane & 15)) * 36 + kd];
      kb1[i] = klds[(16 + (lane & 15)) * 36 + kd];
    }
    floatx8 S0, S1;
    for (int e = 0; e < 8; ++e) { S0[e] = 0.f; S1[e] = 0.f; }
    S0 = __builtin_amdgcn_wmma_f32_16x16x32_bf16(false, qa, false, kb0, (short)0, S0, false, false);
    S1 = __builtin_amdgcn_wmma_f32_16x16x32_bf16(false, qa, false, kb1, (short)0, S1, false, false);

    // online softmax: row max via DPP16 butterfly; row sum via P @ ones WMMA
    for (int j = 0; j < 8; ++j) {
      float vmax = group16_max(fmaxf(S0[j], S1[j]));
      float newm = fmaxf(mrun[j], vmax);
      float corr = __expf(mrun[j] - newm);
      S0[j] = __expf(S0[j] - newm);
      S1[j] = __expf(S1[j] - newm);
      mrun[j] = newm;
      acc0[j] *= corr;
      acc1[j] *= corr;
      accS[j] *= corr;
    }

    // D-layout -> A-layout via wave-private LDS
    bf16* pw = plds[w];
    {
      int half = (lane < 16) ? 0 : 8;
      int c = lane & 15;
      for (int j = 0; j < 8; ++j) {
        pw[(j + half) * 34 + c]      = (bf16)S0[j];
        pw[(j + half) * 34 + 16 + c] = (bf16)S1[j];
      }
    }
    asm volatile("s_wait_dscnt 0" ::: "memory");
    bf16x16 pa;
    for (int i = 0; i < 16; ++i)
      pa[i] = pw[(lane & 15) * 34 + a_kidx(lane, i)];

    bf16x16 vb0, vb1;
    for (int i = 0; i < 16; ++i) {
      int kn = b_kidx(lane, i);
      vb0[i] = vlds[kn * 36 + (lane & 15)];
      vb1[i] = vlds[kn * 36 + 16 + (lane & 15)];
    }
    acc0 = __builtin_amdgcn_wmma_f32_16x16x32_bf16(false, pa, false, vb0, (short)0, acc0, false, false);
    acc1 = __builtin_amdgcn_wmma_f32_16x16x32_bf16(false, pa, false, vb1, (short)0, acc1, false, false);
    accS = __builtin_amdgcn_wmma_f32_16x16x32_bf16(false, pa, false, ones, (short)0, accS, false, false);
    __syncthreads();
  }

  int half = (lane < 16) ? 0 : 8;
  int c = lane & 15;
  if (Ob) {
    // bf16 out (stage 1): OM=32, OD=1 -> lane-contiguous stores already
    for (int j = 0; j < 8; ++j) {
      float inv = 1.f / accS[j];
      int m = m0 + j + half;
      size_t base = (size_t)b * OB + (size_t)h * OH + (size_t)m * OM;
      Ob[base + (size_t)c * OD]        = (bf16)(acc0[j] * inv);
      Ob[base + (size_t)(16 + c) * OD] = (bf16)(acc1[j] * inv);
    }
  } else {
    // f32 out (stage 2): transpose tile in LDS so each lane writes a 64B run
    float* os = ostage[w];
    for (int j = 0; j < 8; ++j) {
      float inv = 1.f / accS[j];
      os[(c)      * 20 + (j + half)] = acc0[j] * inv;
      os[(16 + c) * 20 + (j + half)] = acc1[j] * inv;
    }
    asm volatile("s_wait_dscnt 0" ::: "memory");
    // lane L owns d=L: 16 contiguous m (OM==1), stride OD between d rows
    float* dst = Of + (size_t)b * OB + (size_t)h * OH + (size_t)m0 * OM +
                 (size_t)lane * OD;
    const float4* src = (const float4*)&os[lane * 20];
    float4 o0 = src[0], o1 = src[1], o2 = src[2], o3 = src[3];
    float4* dv = (float4*)dst;
    dv[0] = o0; dv[1] = o1; dv[2] = o2; dv[3] = o3;
  }
}

// ---------------- LePE depthwise 3x3 + permuted add ----------------
// block = 16 channels x 16 pixels of one image row; LDS halo tile
__global__ __launch_bounds__(256) void sgca_lepe_add_kernel(
    const bf16* __restrict__ v, const float* __restrict__ lw,
    const float* __restrict__ lb, float* __restrict__ out,
    int C, int H, int W) {
  __shared__ bf16 tile[16][3][20];  // [c][ky][x halo], 18 used
  int b = blockIdx.x;
  int c0 = blockIdx.y * 16;
  int p0 = blockIdx.z * 16;
  int HW = H * W;
  int y0 = p0 / W, x0 = p0 % W;
  int t = threadIdx.x;
  for (int idx = t; idx < 16 * 3 * 18; idx += 256) {
    int cl = idx / 54;
    int rem = idx % 54;
    int ky = rem / 18;
    int xl = rem % 18;
    int yy = y0 + ky - 1;
    int xx = x0 + xl - 1;
    bf16 val = (bf16)0.f;
    if (yy >= 0 && yy < H && xx >= 0 && xx < W)
      val = v[((size_t)b * C + (c0 + cl)) * HW + yy * W + xx];
    tile[cl][ky][xl] = val;
  }
  __syncthreads();
  int cl = t & 15, pl = t >> 4;  // consecutive threads -> consecutive channels
  int c = c0 + cl;
  float acc = lb[c];
  for (int ky = 0; ky < 3; ++ky)
    for (int kx = 0; kx < 3; ++kx)
      acc += lw[c * 9 + ky * 3 + kx] * (float)tile[cl][ky][pl + kx];
  // faithful permute quirk: [B,C,HW]->[B,HW,C]->view[B,C,H,W]
  out[(size_t)b * C * HW + (size_t)(p0 + pl) * C + c] += acc;
}

extern "C" void kernel_launch(void* const* d_in, const int* in_sizes, int n_in,
                              void* d_out, int out_size, void* d_ws, size_t ws_size,
                              hipStream_t stream) {
  const int B = 4, C = 256, H = 64, W = 64, HW = 4096, NSP = 256, QK = 256;
  const float scale = 0.17677669529663687f;  // (C/h)^-0.5 = 1/sqrt(32)
  const float* x   = (const float*)d_in[0];
  const float* aff = (const float*)d_in[1];
  const float* nw  = (const float*)d_in[2];
  const float* nb  = (const float*)d_in[3];
  const float* Wq  = (const float*)d_in[4];
  const float* Wk  = (const float*)d_in[5];
  const float* Wv  = (const float*)d_in[6];
  const float* Wsp = (const float*)d_in[7];
  const float* lw  = (const float*)d_in[8];
  const float* lb  = (const float*)d_in[9];
  float* out = (float*)d_out;
  (void)in_sizes; (void)n_in; (void)out_size; (void)ws_size;

  char* ws = (char*)d_ws;
  size_t off = 0;
  auto alloc = [&](size_t bytes) {
    void* p = ws + off;
    off = (off + bytes + 255) & ~(size_t)255;
    return p;
  };
  bf16* xn     = (bf16*)alloc((size_t)B * C * HW * 2);
  float* rs    = (float*)alloc((size_t)B * NSP * 4);
  bf16* stoken = (bf16*)alloc((size_t)B * NSP * C * 2);
  bf16* st     = (bf16*)alloc((size_t)B * QK * NSP * 2);
  bf16* q      = (bf16*)alloc((size_t)B * QK * HW * 2);
  bf16* k      = (bf16*)alloc((size_t)B * QK * HW * 2);
  bf16* v      = (bf16*)alloc((size_t)B * C * HW * 2);
  bf16* soutT  = (bf16*)alloc((size_t)B * NHEADS * NSP * DH * 2);

  sgca_layernorm_kernel<<<dim3(B, HW / 256), 256, 0, stream>>>(x, nw, nb, xn, C, HW);
  sgca_rowsum_kernel<<<dim3(B * NSP), 256, 0, stream>>>(aff, rs, HW);

  // stoken[m,c] = (aff[m,:] . xn[c,:]) / (rowsum[m]+1e-16)   M=NSP N=C K=HW, BT
  sgca_gemm_kernel<true, true><<<dim3(NSP / 128, C / 128, B), 256, 0, stream>>>(
      aff, xn, stoken, rs, NSP, HW, HW, HW, C,
      (long)NSP * HW, (long)C * HW, (long)NSP * C);

  // st[qd,m] = Wsp[qd,:] . stoken[m,:]   M=QK N=NSP K=C, BT
  sgca_gemm_kernel<true, false><<<dim3(QK / 128, NSP / 128, B), 256, 0, stream>>>(
      Wsp, stoken, st, nullptr, 0, C, C, C, NSP,
      0L, (long)NSP * C, (long)QK * NSP);

  // q/k/v projections: M=QK N=HW K=C
  sgca_gemm_kernel<false, false><<<dim3(QK / 128, HW / 128, B), 256, 0, stream>>>(
      Wq, xn, q, nullptr, 0, C, C, HW, HW, 0L, (long)C * HW, (long)QK * HW);
  sgca_gemm_kernel<false, false><<<dim3(QK / 128, HW / 128, B), 256, 0, stream>>>(
      Wk, xn, k, nullptr, 0, C, C, HW, HW, 0L, (long)C * HW, (long)QK * HW);
  sgca_gemm_kernel<false, false><<<dim3(QK / 128, HW / 128, B), 256, 0, stream>>>(
      Wv, xn, v, nullptr, 0, C, C, HW, HW, 0L, (long)C * HW, (long)QK * HW);

  // stage 1: s_out^T = FA(Q=st^T [256x32], K=k^T [4096x32], V=v^T [4096x32])
  sgca_flash_kernel<<<dim3(NSP / 128, B * NHEADS), 256, 0, stream>>>(
      st, k, v, soutT, nullptr, HW, scale,
      (long)QK * NSP, (long)DH * NSP, 1, NSP,
      (long)QK * HW,  (long)DH * HW,  1, HW,
      (long)C * HW,   (long)DH * HW,  1, HW,
      (long)NHEADS * NSP * DH, (long)NSP * DH, DH, 1);

  // stage 2: x_out^T = FA(Q=q^T [4096x32], K=st^T [256x32], V=s_out^T [256x32]) -> f32 out
  sgca_flash_kernel<<<dim3(HW / 128, B * NHEADS), 256, 0, stream>>>(
      q, st, soutT, (bf16*)nullptr, out, NSP, scale,
      (long)QK * HW,  (long)DH * HW,  1, HW,
      (long)QK * NSP, (long)DH * NSP, 1, NSP,
      (long)NHEADS * NSP * DH, (long)NSP * DH, DH, 1,
      (long)C * HW,   (long)DH * HW,  1, HW);

  // LePE add with the reference's permute-reshape quirk
  sgca_lepe_add_kernel<<<dim3(B, C / 16, HW / 16), 256, 0, stream>>>(
      v, lw, lb, out, C, H, W);
}
